// MultiHead_24970939859500
// MI455X (gfx1250) — matrix operand order repaired and checked by reading, not compiled, not run on hardware
//
#include <hip/hip_runtime.h>

// ---------------------------------------------------------------------------
// Transformer block (MHA + FFN + 2x LayerNorm) for MI455X / gfx1250, wave32.
// All GEMMs use v_wmma_f32_16x16x32_f16 (f16 in, f32 accumulate).
// FFN/proj GEMMs stage weight tiles in LDS via global_load_async_to_lds_b128
// (ASYNCcnt double-buffered) so the whole block shares each weight fragment.
// ---------------------------------------------------------------------------

typedef __attribute__((ext_vector_type(16))) _Float16 v16h;
typedef __attribute__((ext_vector_type(8)))  _Float16 v8h;
typedef __attribute__((ext_vector_type(8)))  float    v8f;
typedef __attribute__((ext_vector_type(4)))  int      v4i;

static constexpr int B  = 4;
static constexpr int S  = 2048;
static constexpr int E  = 512;
static constexpr int H  = 8;
static constexpr int D  = 64;
static constexpr int BS = B * S;       // 8192 rows
static constexpr int F  = 4 * E;       // 2048 FFN hidden

// ---- gfx1250 async global->LDS copy (ASYNCcnt) with portable fallback ----
#if defined(__has_builtin)
#if __has_builtin(__builtin_amdgcn_global_load_async_to_lds_b128)
#define HAS_ASYNC_LDS 1
#endif
#endif
#ifndef HAS_ASYNC_LDS
#define HAS_ASYNC_LDS 0
#endif

static __device__ __forceinline__ void copy16_g2l(const _Float16* g, _Float16* l) {
#if HAS_ASYNC_LDS
  __builtin_amdgcn_global_load_async_to_lds_b128(
      (__attribute__((address_space(1))) v4i*)(g),
      (__attribute__((address_space(3))) v4i*)(l), 0, 0);
#else
  *(v8h*)l = *(const v8h*)g;
#endif
}

static __device__ __forceinline__ void wait_async_le2() {
#if HAS_ASYNC_LDS
#if __has_builtin(__builtin_amdgcn_s_wait_asynccnt)
  __builtin_amdgcn_s_wait_asynccnt(2);
#else
  asm volatile("s_wait_asynccnt 0x2" ::: "memory");
#endif
#endif
}
static __device__ __forceinline__ void wait_async_0() {
#if HAS_ASYNC_LDS
#if __has_builtin(__builtin_amdgcn_s_wait_asynccnt)
  __builtin_amdgcn_s_wait_asynccnt(0);
#else
  asm volatile("s_wait_asynccnt 0x0" ::: "memory");
#endif
#endif
}

static __device__ __forceinline__ v8f wmma_f16(v16h a, v16h b, v8f c) {
  return __builtin_amdgcn_wmma_f32_16x16x32_f16(false, a, false, b,
                                                (short)0, c, false, false);
}

// A-matrix fragment (16x32 f16), this code's k order (not ISA table order): lane m = lane%16,
// halves 0..15 <- K = k0..k0+15 (contiguous, matches B fragment),
// k0 = lane<16 ? 0 : 16.   Two 16-byte loads per lane.
static __device__ __forceinline__ v16h load_a_frag(const _Float16* base, int ld) {
  const int lane = threadIdx.x & 31;
  const int m  = lane & 15;
  const int k0 = (lane & 16) ? 16 : 0;
  const _Float16* r = base + (size_t)m * ld + k0;
  v8h lo = *(const v8h*)(r);
  v8h hi = *(const v8h*)(r + 8);
  return __builtin_shufflevector(lo, hi, 0,1,2,3,4,5,6,7,8,9,10,11,12,13,14,15);
}

// B-matrix fragment (32x16 f16): lane n = lane%16 holds column n,
// K = 0..15 (lanes 0-15) or 16..31 (lanes 16-31), contiguous: one 32B load.
// `base` points at an N-major (N rows x K cols) matrix (i.e. B transposed).
static __device__ __forceinline__ v16h load_b_frag(const _Float16* base, int ld) {
  const int lane = threadIdx.x & 31;
  const int n  = lane & 15;
  const int k0 = (lane & 16) ? 16 : 0;
  return *(const v16h*)(base + (size_t)n * ld + k0);
}

static __device__ __forceinline__ v8f vzero8() {
  v8f z = {0.f,0.f,0.f,0.f,0.f,0.f,0.f,0.f};
  return z;
}

// ---------------------------------------------------------------------------
// Prep: f32 -> f16 convert, and batched transpose-convert (K,N) -> (N,K).
// ---------------------------------------------------------------------------
__global__ __launch_bounds__(256)
void cvt_f16_kernel(const float* __restrict__ in, _Float16* __restrict__ out, int n) {
  int i = blockIdx.x * 256 + threadIdx.x;
  if (i < n) out[i] = (_Float16)in[i];
}

__global__ __launch_bounds__(256)
void transpose_cvt_kernel(const float* __restrict__ in, _Float16* __restrict__ out,
                          int batch, int K, int N) {
  int i = blockIdx.x * 256 + threadIdx.x;
  int total = batch * K * N;
  if (i >= total) return;
  int b   = i / (K * N);
  int rem = i - b * (K * N);
  int n   = rem / K;
  int k   = rem - n * K;
  out[i] = (_Float16)in[(size_t)b * K * N + (size_t)k * N + n];
}

// ---------------------------------------------------------------------------
// QKV projection: per (b,h), per 16-row s-tile (one wave), 16x64 output.
// Q is scaled by 1/sqrt(D)=0.125 (exact in f16). V is stored transposed (D,S).
// ---------------------------------------------------------------------------
__global__ __launch_bounds__(128)
void qkv_gemm_kernel(const _Float16* __restrict__ x16,
                     const _Float16* __restrict__ WqT,  // (H, D, E)
                     const _Float16* __restrict__ WkT,
                     const _Float16* __restrict__ WvT,
                     _Float16* __restrict__ q16,        // (B,H,S,D)
                     _Float16* __restrict__ k16,        // (B,H,S,D)
                     _Float16* __restrict__ vT16) {     // (B,H,D,S)
  const int lane = threadIdx.x & 31;
  const int wave = threadIdx.x >> 5;
  const int s0   = (blockIdx.x * 4 + wave) * 16;
  const int bh   = blockIdx.y;                // b*H + h
  const int b    = bh >> 3;
  const int h    = bh & 7;
  const int mat  = blockIdx.z;                // 0=Q 1=K 2=V
  const _Float16* WT = (mat == 0) ? WqT : (mat == 1) ? WkT : WvT;
  const _Float16* A  = x16 + (size_t)(b * S + s0) * E;
  const _Float16* Wh = WT + (size_t)h * D * E;

  v8f acc[4];
#pragma unroll
  for (int i = 0; i < 4; ++i) acc[i] = vzero8();

  for (int kk = 0; kk < E; kk += 32) {
    v16h a = load_a_frag(A + kk, E);
#pragma unroll
    for (int dt = 0; dt < 4; ++dt) {
      v16h bf = load_b_frag(Wh + (size_t)(dt * 16) * E + kk, E);
      acc[dt] = wmma_f16(a, bf, acc[dt]);
    }
  }

  const int col   = lane & 15;
  const int rbase = (lane & 16) ? 8 : 0;
  if (mat == 2) {
    _Float16* Vt = vT16 + (size_t)bh * D * S;
#pragma unroll
    for (int dt = 0; dt < 4; ++dt)
#pragma unroll
      for (int g = 0; g < 8; ++g)
        Vt[(size_t)(dt * 16 + col) * S + s0 + rbase + g] = (_Float16)acc[dt][g];
  } else {
    _Float16* O = ((mat == 0) ? q16 : k16) + (size_t)(bh * S + s0) * D;
    const float scale = (mat == 0) ? 0.125f : 1.0f;
#pragma unroll
    for (int dt = 0; dt < 4; ++dt)
#pragma unroll
      for (int g = 0; g < 8; ++g)
        O[(size_t)(rbase + g) * D + dt * 16 + col] = (_Float16)(acc[dt][g] * scale);
  }
}

// ---------------------------------------------------------------------------
// Causal flash attention: one wave owns 16 query rows, streams keys in
// blocks of 32, online softmax, P->A-frag transpose through LDS.
// ---------------------------------------------------------------------------
__global__ __launch_bounds__(128)
void attn_kernel(const _Float16* __restrict__ q16,   // (B,H,S,D), pre-scaled
                 const _Float16* __restrict__ k16,   // (B,H,S,D)
                 const _Float16* __restrict__ vT16,  // (B,H,D,S)
                 _Float16* __restrict__ o16) {       // (B,S,H*D)
  __shared__ __align__(32) _Float16 lds[4][16 * 32];
  const int lane = threadIdx.x & 31;
  const int wave = threadIdx.x >> 5;
  const int qs   = (blockIdx.x * 4 + wave) * 16;
  const int bh   = blockIdx.y;
  const int b    = bh >> 3;
  const int h    = bh & 7;

  const _Float16* Q  = q16 + (size_t)(bh * S + qs) * D;
  const _Float16* K  = k16 + (size_t)bh * S * D;
  const _Float16* Vt = vT16 + (size_t)bh * D * S;

  const v16h a0 = load_a_frag(Q, D);        // d = 0..31
  const v16h a1 = load_a_frag(Q + 32, D);   // d = 32..63

  v8f acc[4];
#pragma unroll
  for (int i = 0; i < 4; ++i) acc[i] = vzero8();
  float mrow[8], lrow[8];
#pragma unroll
  for (int g = 0; g < 8; ++g) { mrow[g] = -3.0e38f; lrow[g] = 0.0f; }

  const int   col   = lane & 15;
  const int   rbase = (lane & 16) ? 8 : 0;
  const float LOG2E = 1.44269504f;
  const int   nk    = qs + 16;              // keys needed by last row

  for (int t0 = 0; t0 < nk; t0 += 32) {
    // ---- scores: two 16x16 tiles (keys t0..t0+15, t0+16..t0+31) ----
    v8f s0 = vzero8(), s1 = vzero8();
    {
      v16h kb;
      kb = load_b_frag(K + (size_t)t0 * D, D);              s0 = wmma_f16(a0, kb, s0);
      kb = load_b_frag(K + (size_t)t0 * D + 32, D);         s0 = wmma_f16(a1, kb, s0);
      kb = load_b_frag(K + (size_t)(t0 + 16) * D, D);       s1 = wmma_f16(a0, kb, s1);
      kb = load_b_frag(K + (size_t)(t0 + 16) * D + 32, D);  s1 = wmma_f16(a1, kb, s1);
    }
    // ---- causal mask + online softmax ----
#pragma unroll
    for (int g = 0; g < 8; ++g) {
      const int srow = qs + rbase + g;
      float e0 = ((t0 + col)      <= srow) ? (float)s0[g] : -3.0e38f;
      float e1 = ((t0 + 16 + col) <= srow) ? (float)s1[g] : -3.0e38f;
      float mx = fmaxf(e0, e1);
#pragma unroll
      for (int off = 1; off < 16; off <<= 1)      // stays in 16-lane group
        mx = fmaxf(mx, __shfl_xor(mx, off, 32));
      mx = fmaxf(mx, mrow[g]);
      const float al = exp2f((mrow[g] - mx) * LOG2E);
      const float p0 = exp2f((e0 - mx) * LOG2E);
      const float p1 = exp2f((e1 - mx) * LOG2E);
      float ps = p0 + p1;
#pragma unroll
      for (int off = 1; off < 16; off <<= 1)
        ps += __shfl_xor(ps, off, 32);
      lrow[g] = lrow[g] * al + ps;
      mrow[g] = mx;
      s0[g] = p0;
      s1[g] = p1;
#pragma unroll
      for (int dt = 0; dt < 4; ++dt) acc[dt][g] *= al;   // rescale O
    }
    // ---- P (C-layout) -> LDS -> A-fragment (16x32) ----
    _Float16* P = lds[wave];
#pragma unroll
    for (int g = 0; g < 8; ++g) {
      P[(rbase + g) * 32 + col]      = (_Float16)s0[g];
      P[(rbase + g) * 32 + 16 + col] = (_Float16)s1[g];
    }
    asm volatile("s_wait_dscnt 0x0" ::: "memory");
    const v16h pa = load_a_frag(P, 32);
    asm volatile("s_wait_dscnt 0x0" ::: "memory");
    // ---- O += P * V ----
#pragma unroll
    for (int dt = 0; dt < 4; ++dt) {
      v16h vb = load_b_frag(Vt + (size_t)(dt * 16) * S + t0, S);
      acc[dt] = wmma_f16(pa, vb, acc[dt]);
    }
  }

  // ---- normalize and write concat-head output ----
  _Float16* O = o16 + (size_t)(b * S + qs) * (H * D) + h * D;
#pragma unroll
  for (int g = 0; g < 8; ++g) {
    const float inv = 1.0f / lrow[g];
#pragma unroll
    for (int dt = 0; dt < 4; ++dt)
      O[(size_t)(rbase + g) * (H * D) + dt * 16 + col] = (_Float16)(acc[dt][g] * inv);
  }
}

// ---------------------------------------------------------------------------
// Generic WMMA GEMM: Y(M,N) = X(M,K) * W, W given transposed as WT(N,K).
// Block tile: 128 rows x 64 cols (4 waves x 32 rows). The 64x32 weight
// K-slice is staged once per block into LDS with async global->LDS copies,
// double-buffered across K-steps. 8 WMMAs per wave per K-step.
// MODE 0: write f32 (no bias)    [out projection]
// MODE 1: +bias, ReLU, write f16 [FFN layer 1]
// MODE 2: +bias, write f32       [FFN layer 2]
// ---------------------------------------------------------------------------
template <int MODE>
__global__ __launch_bounds__(128)
void gemm_kernel(const _Float16* __restrict__ X,
                 const _Float16* __restrict__ WT,
                 const float* __restrict__ bias,
                 float* __restrict__ outF,
                 _Float16* __restrict__ outH,
                 int N, int Kdim) {
  __shared__ __align__(32) _Float16 ldsB[2][64 * 32];   // 2 x 4KB double buffer
  const int lane = threadIdx.x & 31;
  const int wave = threadIdx.x >> 5;
  const int tid  = threadIdx.x;
  const int row0 = blockIdx.x * 128 + wave * 32;
  const int n0   = blockIdx.y * 64;
  const int nsteps = Kdim / 32;

  // Stage the 64x32 weight slice for K-step `step` into buffer `buf`.
  // 4KB = 256 x 16B chunks; 2 chunks per thread. chunk c -> (n = c>>2, k8 = (c&3)*8)
  auto prefetch = [&](int step, int buf) {
    const int kk = step * 32;
#pragma unroll
    for (int j = 0; j < 2; ++j) {
      const int c  = tid + j * 128;
      const int n  = c >> 2;
      const int k8 = (c & 3) * 8;
      copy16_g2l(WT + (size_t)(n0 + n) * Kdim + kk + k8,
                 &ldsB[buf][n * 32 + k8]);
    }
  };

  v8f acc[2][4];
#pragma unroll
  for (int r = 0; r < 2; ++r)
#pragma unroll
    for (int i = 0; i < 4; ++i) acc[r][i] = vzero8();

  const _Float16* A0 = X + (size_t)row0 * Kdim;

  prefetch(0, 0);
  for (int i = 0; i < nsteps; ++i) {
    const int  cur     = i & 1;
    const bool hasNext = (i + 1) < nsteps;
    if (hasNext) {
      prefetch(i + 1, cur ^ 1);
      wait_async_le2();       // in-order completion => buffer `cur` landed
    } else {
      wait_async_0();
    }
    __syncthreads();          // publish LDS writes across waves

    const int kk = i * 32;
    const v16h a0 = load_a_frag(A0 + kk, Kdim);
    const v16h a1 = load_a_frag(A0 + (size_t)16 * Kdim + kk, Kdim);
#pragma unroll
    for (int dt = 0; dt < 4; ++dt) {
      v16h bf = load_b_frag(&ldsB[cur][dt * 16 * 32], 32);
      acc[0][dt] = wmma_f16(a0, bf, acc[0][dt]);
      acc[1][dt] = wmma_f16(a1, bf, acc[1][dt]);
    }
    __syncthreads();          // all reads of `cur` done before it is refilled
  }

  const int col   = lane & 15;
  const int rbase = (lane & 16) ? 8 : 0;
#pragma unroll
  for (int rt = 0; rt < 2; ++rt) {
#pragma unroll
    for (int dt = 0; dt < 4; ++dt) {
      const int n = n0 + dt * 16 + col;
#pragma unroll
      for (int g = 0; g < 8; ++g) {
        float v = acc[rt][dt][g];
        if (MODE >= 1) v += bias[n];
        const size_t idx = (size_t)(row0 + rt * 16 + rbase + g) * N + n;
        if (MODE == 1) {
          v = fmaxf(v, 0.0f);
          outH[idx] = (_Float16)v;
        } else {
          outF[idx] = v;
        }
      }
    }
  }
}

// ---------------------------------------------------------------------------
// Residual + LayerNorm over E=512: one wave per row (32 lanes x 16 elems).
// OUT_HALF=1 -> f16 output (feeds FFN GEMM); else f32 output.
// ---------------------------------------------------------------------------
template <int OUT_HALF>
__global__ __launch_bounds__(256)
void ln_kernel(const float* __restrict__ a, const float* __restrict__ r,
               const float* __restrict__ g, const float* __restrict__ bb,
               float* __restrict__ outF, _Float16* __restrict__ outH) {
  const int lane = threadIdx.x & 31;
  const int wave = threadIdx.x >> 5;
  const int row  = blockIdx.x * 8 + wave;
  const float* pa = a + (size_t)row * E;
  const float* pr = r + (size_t)row * E;

  float vals[16];
  float s1 = 0.0f, s2 = 0.0f;
#pragma unroll
  for (int i = 0; i < 16; ++i) {
    const float t = pa[lane + i * 32] + pr[lane + i * 32];
    vals[i] = t;
    s1 += t;
    s2 += t * t;
  }
#pragma unroll
  for (int off = 1; off < 32; off <<= 1) {
    s1 += __shfl_xor(s1, off, 32);
    s2 += __shfl_xor(s2, off, 32);
  }
  const float mean = s1 * (1.0f / E);
  const float var  = s2 * (1.0f / E) - mean * mean;
  const float rinv = rsqrtf(var + 1e-5f);
#pragma unroll
  for (int i = 0; i < 16; ++i) {
    const int c = lane + i * 32;
    const float y = (vals[i] - mean) * rinv * g[c] + bb[c];
    if (OUT_HALF) outH[(size_t)row * E + c] = (_Float16)y;
    else          outF[(size_t)row * E + c] = y;
  }
}

// ---------------------------------------------------------------------------
// Host orchestration
// ---------------------------------------------------------------------------
extern "C" void kernel_launch(void* const* d_in, const int* in_sizes, int n_in,
                              void* d_out, int out_size, void* d_ws, size_t ws_size,
                              hipStream_t stream) {
  const float* x     = (const float*)d_in[0];   // (B,S,E)
  const float* Wq    = (const float*)d_in[1];   // (H,E,D)
  const float* Wk    = (const float*)d_in[2];
  const float* Wv    = (const float*)d_in[3];
  const float* Wproj = (const float*)d_in[4];   // (H*D,E)
  const float* g1    = (const float*)d_in[5];
  const float* b1    = (const float*)d_in[6];
  const float* W1    = (const float*)d_in[7];   // (E,4E)
  const float* bb1   = (const float*)d_in[8];
  const float* W2    = (const float*)d_in[9];   // (4E,E)
  const float* bb2   = (const float*)d_in[10];
  const float* g2    = (const float*)d_in[11];
  const float* b2    = (const float*)d_in[12];
  float* out = (float*)d_out;

  // workspace allocator (256B aligned segments)
  char* wp = (char*)d_ws;
  auto alloc = [&](size_t bytes) -> void* {
    void* p = (void*)wp;
    wp += (bytes + 255) & ~(size_t)255;
    return p;
  };
  _Float16* x16    = (_Float16*)alloc((size_t)BS * E * 2);
  _Float16* WqT    = (_Float16*)alloc((size_t)H * D * E * 2);   // (H,D,E)
  _Float16* WkT    = (_Float16*)alloc((size_t)H * D * E * 2);
  _Float16* WvT    = (_Float16*)alloc((size_t)H * D * E * 2);
  _Float16* WprojT = (_Float16*)alloc((size_t)E * (H * D) * 2); // (E, H*D)
  _Float16* W1T    = (_Float16*)alloc((size_t)F * E * 2);       // (4E, E)
  _Float16* W2T    = (_Float16*)alloc((size_t)E * F * 2);       // (E, 4E)
  _Float16* q16    = (_Float16*)alloc((size_t)B * H * S * D * 2);
  _Float16* k16    = (_Float16*)alloc((size_t)B * H * S * D * 2);
  _Float16* vT16   = (_Float16*)alloc((size_t)B * H * D * S * 2);
  _Float16* o16    = (_Float16*)alloc((size_t)BS * (H * D) * 2);
  float*    yproj  = (float*)   alloc((size_t)BS * E * 4);
  _Float16* y1h    = (_Float16*)alloc((size_t)BS * E * 2);
  _Float16* hid16  = (_Float16*)alloc((size_t)BS * F * 2);
  float*    f2     = (float*)   alloc((size_t)BS * E * 4);

  // ---- prep: convert x, transpose+convert weights ----
  {
    int n = BS * E;
    cvt_f16_kernel<<<(n + 255) / 256, 256, 0, stream>>>(x, x16, n);
  }
  {
    int n = H * E * D;
    transpose_cvt_kernel<<<(n + 255) / 256, 256, 0, stream>>>(Wq, WqT, H, E, D);
    transpose_cvt_kernel<<<(n + 255) / 256, 256, 0, stream>>>(Wk, WkT, H, E, D);
    transpose_cvt_kernel<<<(n + 255) / 256, 256, 0, stream>>>(Wv, WvT, H, E, D);
  }
  {
    int n = (H * D) * E;
    transpose_cvt_kernel<<<(n + 255) / 256, 256, 0, stream>>>(Wproj, WprojT, 1, H * D, E);
  }
  {
    int n = E * F;
    transpose_cvt_kernel<<<(n + 255) / 256, 256, 0, stream>>>(W1, W1T, 1, E, F);
    transpose_cvt_kernel<<<(n + 255) / 256, 256, 0, stream>>>(W2, W2T, 1, F, E);
  }

  // ---- QKV projections ----
  qkv_gemm_kernel<<<dim3(S / 64, B * H, 3), 128, 0, stream>>>(
      x16, WqT, WkT, WvT, q16, k16, vT16);

  // ---- causal flash attention ----
  attn_kernel<<<dim3(S / 64, B * H), 128, 0, stream>>>(q16, k16, vT16, o16);

  // ---- output projection ----
  gemm_kernel<0><<<dim3(BS / 128, E / 64), 128, 0, stream>>>(
      o16, WprojT, nullptr, yproj, nullptr, E, H * D);

  // ---- LN1: layernorm(x + proj) -> f16 for FFN ----
  ln_kernel<1><<<BS / 8, 256, 0, stream>>>(yproj, x, g1, b1, nullptr, y1h);

  // ---- FFN layer 1: relu(y1 @ W1 + bb1) -> f16 hidden ----
  gemm_kernel<1><<<dim3(BS / 128, F / 64), 128, 0, stream>>>(
      y1h, W1T, bb1, nullptr, hid16, F, E);

  // ---- FFN layer 2: hidden @ W2 + bb2 -> f32 ----
  gemm_kernel<2><<<dim3(BS / 128, E / 64), 128, 0, stream>>>(
      hid16, W2T, bb2, f2, nullptr, E, F);

  // ---- LN2: layernorm(ff + x) -> f32 output ----
  ln_kernel<0><<<BS / 8, 256, 0, stream>>>(f2, x, g2, b2, out, nullptr);

  (void)in_sizes; (void)n_in; (void)out_size; (void)ws_size;
}
